// single_MogLSTM_24034636988657
// MI455X (gfx1250) — matrix operand order repaired and checked by reading, not compiled, not run on hardware
//
#include <hip/hip_runtime.h>
#include <hip/hip_bf16.h>

// ---------------------------------------------------------------------------
// Mogrifier LSTM, persistent-kernel implementation for gfx1250 (MI455X).
// B=256, S=512, C=H=512. One persistent kernel runs the whole recurrence;
// software grid barrier between dependent GEMM phases. All GEMMs use
// v_wmma_f32_16x16x32_bf16 with f32 accumulation. Weights are packed once
// per launch into the per-lane WMMA B-operand layout (bf16).
// ---------------------------------------------------------------------------

#define Bsz 256
#define Ssz 512
#define Csz 512
#define Hsz 512
#define NBLK 16
#define NTHR 256
#define WPB  (NTHR / 32)
#define NWAVES (NBLK * WPB)   // 128 waves

typedef __attribute__((ext_vector_type(16))) __bf16 v16bf;
typedef __attribute__((ext_vector_type(8)))  __bf16 v8bf;
typedef __attribute__((ext_vector_type(8)))  float  v8f;

// ------------------------- workspace layout (bytes) ------------------------
constexpr size_t OFF_CNT = 0;                          // grid-barrier counter
constexpr size_t OFF_XT  = 256;                        // fp32 B x C
constexpr size_t OFF_HT  = OFF_XT  + (size_t)Bsz*Csz*4;
constexpr size_t OFF_CT  = OFF_HT  + (size_t)Bsz*Hsz*4;
constexpr size_t OFF_MT  = OFF_CT  + (size_t)Bsz*Hsz*4;
constexpr size_t OFF_GT  = OFF_MT  + (size_t)Bsz*Hsz*4;      // fp32 B x 4H
constexpr size_t OFF_XTB = OFF_GT  + (size_t)Bsz*4*Hsz*4;    // bf16 B x C
constexpr size_t OFF_HTB = OFF_XTB + (size_t)Bsz*Csz*2;
constexpr size_t OFF_MTB = OFF_HTB + (size_t)Bsz*Hsz*2;
constexpr size_t OFF_BM  = OFF_MTB + (size_t)Bsz*Hsz*2;      // fp32 H
constexpr size_t OFF_BG  = OFF_BM  + (size_t)Hsz*4;          // fp32 4H
constexpr size_t OFF_QP  = OFF_BG  + (size_t)4*Hsz*4;        // packed bf16 K=512,N=512
constexpr size_t OFF_RP  = OFF_QP  + (size_t)Hsz*Csz*2;
constexpr size_t OFF_MP  = OFF_RP  + (size_t)Csz*Hsz*2;      // K=1024,N=512
constexpr size_t OFF_GP  = OFF_MP  + (size_t)(Csz+Hsz)*Hsz*2; // K=1024,N=2048

// ------------------------------- helpers -----------------------------------
__device__ __forceinline__ unsigned short f2bf(float f) {
  unsigned u = __float_as_uint(f);
  u += 0x7FFFu + ((u >> 16) & 1u);          // round-to-nearest-even
  return (unsigned short)(u >> 16);
}
__device__ __forceinline__ float sigm(float x) {
  return 1.0f / (1.0f + __expf(-x));
}

__device__ __forceinline__ void grid_sync(unsigned* cnt, unsigned target) {
  __threadfence();
  __syncthreads();
  if (threadIdx.x == 0) {
    __hip_atomic_fetch_add(cnt, 1u, __ATOMIC_ACQ_REL, __HIP_MEMORY_SCOPE_AGENT);
    while (__hip_atomic_load(cnt, __ATOMIC_ACQUIRE, __HIP_MEMORY_SCOPE_AGENT) < target)
      __builtin_amdgcn_s_sleep(1);
  }
  __syncthreads();
  __threadfence();
}

// ----------------------------------------------------------------------------
// GEMM phase: D[B x N] = A[B x K] @ Bp  (A bf16 row-major, rows of length 512
// per segment; K may be the concat of two 512-wide segments A0|A1).
// Bp is pre-packed per-lane B layout: index ((nb*KB + kb)*32 + lane)*16 + e,
// lane = (n&15) | (khalf<<4), e = k&15 within a 32-K block.
// MODE 0: D = 2*sigmoid(acc) * src   (mogrifier rounds; src read elementwise)
// MODE 1: D = acc + bias[n]
// Each wave computes a 16x64 strip (4 accumulators reuse one A tile).
// ----------------------------------------------------------------------------
template <int MODE>
__device__ __forceinline__ void gemm_phase(
    const unsigned short* __restrict__ A0, const unsigned short* __restrict__ A1,
    int kbA0, int KB,
    const unsigned short* __restrict__ Bp, int N,
    const float* __restrict__ src, size_t srcLd,
    const float* __restrict__ bias,
    float* __restrict__ D32, unsigned short* __restrict__ D16,
    int gw)
{
  const int lane = threadIdx.x & 31;
  const int lo = lane & 15;
  const int hi = lane >> 4;
  const int nstrips = (Bsz / 16) * (N / 64);
  for (int s = gw; s < nstrips; s += NWAVES) {
    const int m0 = (s / (N / 64)) * 16;
    const int n0 = (s % (N / 64)) * 64;
    v8f acc0 = {}, acc1 = {}, acc2 = {}, acc3 = {};
    const v16bf* bbase = (const v16bf*)Bp + (size_t)(n0 >> 4) * KB * 32;
#pragma unroll 4
    for (int kb = 0; kb < KB; ++kb) {
      const unsigned short* Aseg = (kb < kbA0) ? A0 : A1;
      const int kk = (kb < kbA0) ? kb : (kb - kbA0);
      const unsigned short* ap = Aseg + (size_t)(m0 + lo) * 512 + kk * 32 + hi * 16;
      v8bf a_lo = *(const v8bf*)(ap);
      v8bf a_hi = *(const v8bf*)(ap + 8);
      v16bf a = __builtin_shufflevector(a_lo, a_hi,
                                        0, 1, 2, 3, 4, 5, 6, 7,
                                        8, 9, 10, 11, 12, 13, 14, 15);
      const v16bf* bp = bbase + (size_t)kb * 32 + lane;
      v16bf b0 = bp[0];
      v16bf b1 = bp[(size_t)KB * 32];
      v16bf b2 = bp[(size_t)KB * 32 * 2];
      v16bf b3 = bp[(size_t)KB * 32 * 3];
      acc0 = __builtin_amdgcn_wmma_f32_16x16x32_bf16(false, a, false, b0, (short)0, acc0, false, false);
      acc1 = __builtin_amdgcn_wmma_f32_16x16x32_bf16(false, a, false, b1, (short)0, acc1, false, false);
      acc2 = __builtin_amdgcn_wmma_f32_16x16x32_bf16(false, a, false, b2, (short)0, acc2, false, false);
      acc3 = __builtin_amdgcn_wmma_f32_16x16x32_bf16(false, a, false, b3, (short)0, acc3, false, false);
    }
    v8f accs[4] = {acc0, acc1, acc2, acc3};
#pragma unroll
    for (int j = 0; j < 4; ++j) {
      const int n = n0 + j * 16 + lo;
#pragma unroll
      for (int r = 0; r < 8; ++r) {
        const int m = m0 + hi * 8 + r;
        float v = accs[j][r];
        float o;
        if (MODE == 0) {
          o = 2.0f * sigm(v) * src[(size_t)m * srcLd + n];
        } else {
          o = v + bias[n];
        }
        D32[(size_t)m * N + n] = o;
        if (D16) D16[(size_t)m * N + n] = f2bf(o);
      }
    }
  }
}

// --------------------------- persistent kernel ------------------------------
__global__ void __launch_bounds__(NTHR)
moglstm_main(const float* __restrict__ x, float* __restrict__ out, char* __restrict__ ws)
{
  unsigned* cnt = (unsigned*)(ws + OFF_CNT);
  float* XT = (float*)(ws + OFF_XT);
  float* HT = (float*)(ws + OFF_HT);
  float* CT = (float*)(ws + OFF_CT);
  float* MT = (float*)(ws + OFF_MT);
  float* GT = (float*)(ws + OFF_GT);
  unsigned short* XTb = (unsigned short*)(ws + OFF_XTB);
  unsigned short* HTb = (unsigned short*)(ws + OFF_HTB);
  unsigned short* MTb = (unsigned short*)(ws + OFF_MTB);
  const float* BM = (const float*)(ws + OFF_BM);
  const float* BG = (const float*)(ws + OFF_BG);
  const unsigned short* Qp = (const unsigned short*)(ws + OFF_QP);
  const unsigned short* Rp = (const unsigned short*)(ws + OFF_RP);
  const unsigned short* Mp = (const unsigned short*)(ws + OFF_MP);
  const unsigned short* Gp = (const unsigned short*)(ws + OFF_GP);

  const int gw = blockIdx.x * WPB + (threadIdx.x >> 5);
  unsigned bar = 0;

#pragma unroll 1
  for (int t = 0; t < Ssz; ++t) {
    // P1: xt = 2*sig(ht @ Q) * x[:,t,:]
    gemm_phase<0>(HTb, HTb, 16, 16, Qp, Csz, x + (size_t)t * Csz, (size_t)Ssz * Csz,
                  nullptr, XT, XTb, gw);
    grid_sync(cnt, ++bar * NBLK);
    // P2: ht = 2*sig(xt @ R) * ht
    gemm_phase<0>(XTb, XTb, 16, 16, Rp, Hsz, HT, Hsz, nullptr, HT, HTb, gw);
    grid_sync(cnt, ++bar * NBLK);
    // P3: xt = 2*sig(ht @ Q) * xt
    gemm_phase<0>(HTb, HTb, 16, 16, Qp, Csz, XT, Csz, nullptr, XT, XTb, gw);
    grid_sync(cnt, ++bar * NBLK);
    // P4: ht = 2*sig(xt @ R) * ht
    gemm_phase<0>(XTb, XTb, 16, 16, Rp, Hsz, HT, Hsz, nullptr, HT, HTb, gw);
    grid_sync(cnt, ++bar * NBLK);
    // P5: xt = 2*sig(ht @ Q) * xt
    gemm_phase<0>(HTb, HTb, 16, 16, Qp, Csz, XT, Csz, nullptr, XT, XTb, gw);
    grid_sync(cnt, ++bar * NBLK);
    // P6: mt = [xt|ht] @ Mp + (bmx+bmh)
    gemm_phase<1>(XTb, HTb, 16, 32, Mp, Hsz, nullptr, 0, BM, MT, MTb, gw);
    grid_sync(cnt, ++bar * NBLK);
    // P7: gates = [xt|mt] @ Gp + (bih+bhm)
    gemm_phase<1>(XTb, MTb, 16, 32, Gp, 4 * Hsz, nullptr, 0, BG, GT, nullptr, gw);
    grid_sync(cnt, ++bar * NBLK);
    // P8: LSTM cell elementwise, write out[:,t,:]
    {
      const int tid = blockIdx.x * NTHR + threadIdx.x;
      for (int i = tid; i < Bsz * Hsz; i += NBLK * NTHR) {
        const int b = i >> 9;
        const int h = i & 511;
        const float* g = GT + (size_t)b * (4 * Hsz);
        float ig = sigm(g[h]);
        float fg = sigm(g[Hsz + h]);
        float gg = tanhf(g[2 * Hsz + h]);
        float og = sigm(g[3 * Hsz + h]);
        float c = fg * CT[i] + ig * gg;
        float hn = og * tanhf(c);
        CT[i] = c;
        HT[i] = hn;
        HTb[i] = f2bf(hn);
        out[((size_t)b * Ssz + t) * Hsz + h] = hn;
      }
    }
    grid_sync(cnt, ++bar * NBLK);
  }
}

// ------------------------------ init kernels --------------------------------
__global__ void init_state(char* ws, const float* Wih_b, const float* Whm_b,
                           const float* Wmx_b, const float* Wmh_b)
{
  float* HT = (float*)(ws + OFF_HT);
  float* CT = (float*)(ws + OFF_CT);
  unsigned short* HTb = (unsigned short*)(ws + OFF_HTB);
  float* BM = (float*)(ws + OFF_BM);
  float* BG = (float*)(ws + OFF_BG);
  int i = blockIdx.x * blockDim.x + threadIdx.x;
  if (i < Bsz * Hsz) { HT[i] = 0.f; CT[i] = 0.f; HTb[i] = 0; }
  if (i < 4 * Hsz)   BG[i] = Wih_b[i] + Whm_b[i];
  if (i < Hsz)       BM[i] = Wmx_b[i] + Wmh_b[i];
  if (i == 0)        *(unsigned*)(ws + OFF_CNT) = 0u;
}

// Pack a logical K x N weight (Wt[k][n]) into the per-lane WMMA B layout.
// mode 0: Wt[k][n] = S0[k*N + n]                                  (Q, R)
// mode 1: Wt[k][n] = k<Ksplit ? S0[n*ld0 + k] : S1[n*ld1 + k-Ksplit] (W.T concat)
__global__ void pack_b(const float* __restrict__ S0, const float* __restrict__ S1,
                       unsigned short* __restrict__ Bp,
                       int K, int N, int mode, int Ksplit, int ld0, int ld1)
{
  int idx = blockIdx.x * blockDim.x + threadIdx.x;
  if (idx >= K * N) return;
  const int k = idx / N;
  const int n = idx % N;
  float v;
  if (mode == 0) v = S0[(size_t)k * N + n];
  else           v = (k < Ksplit) ? S0[(size_t)n * ld0 + k]
                                  : S1[(size_t)n * ld1 + (k - Ksplit)];
  const int KB = K >> 5;
  const int nb = n >> 4;
  const int kb = k >> 5;
  const int lane = (n & 15) | (((k >> 4) & 1) << 4);
  const int e = k & 15;
  Bp[((size_t)(nb * KB + kb) * 32 + lane) * 16 + e] = f2bf(v);
}

// -------------------------------- launcher ----------------------------------
extern "C" void kernel_launch(void* const* d_in, const int* in_sizes, int n_in,
                              void* d_out, int out_size, void* d_ws, size_t ws_size,
                              hipStream_t stream)
{
  (void)in_sizes; (void)n_in; (void)out_size; (void)ws_size;
  const float* x     = (const float*)d_in[0];
  const float* Wih_w = (const float*)d_in[1];
  const float* Wih_b = (const float*)d_in[2];
  const float* Wmx_w = (const float*)d_in[3];
  const float* Wmx_b = (const float*)d_in[4];
  const float* Wmh_w = (const float*)d_in[5];
  const float* Wmh_b = (const float*)d_in[6];
  const float* Whm_w = (const float*)d_in[7];
  const float* Whm_b = (const float*)d_in[8];
  const float* Q     = (const float*)d_in[9];
  const float* R     = (const float*)d_in[10];

  char* ws = (char*)d_ws;
  unsigned short* Qp = (unsigned short*)(ws + OFF_QP);
  unsigned short* Rp = (unsigned short*)(ws + OFF_RP);
  unsigned short* Mp = (unsigned short*)(ws + OFF_MP);
  unsigned short* Gp = (unsigned short*)(ws + OFF_GP);

  init_state<<<(Bsz * Hsz + NTHR - 1) / NTHR, NTHR, 0, stream>>>(ws, Wih_b, Whm_b, Wmx_b, Wmh_b);

  pack_b<<<(Hsz * Csz + NTHR - 1) / NTHR, NTHR, 0, stream>>>(Q, Q, Qp, Hsz, Csz, 0, 0, 0, 0);
  pack_b<<<(Csz * Hsz + NTHR - 1) / NTHR, NTHR, 0, stream>>>(R, R, Rp, Csz, Hsz, 0, 0, 0, 0);
  pack_b<<<((Csz + Hsz) * Hsz + NTHR - 1) / NTHR, NTHR, 0, stream>>>(
      Wmx_w, Wmh_w, Mp, Csz + Hsz, Hsz, 1, Csz, Csz, Hsz);
  pack_b<<<((Csz + Hsz) * 4 * Hsz + NTHR - 1) / NTHR, NTHR, 0, stream>>>(
      Wih_w, Whm_w, Gp, Csz + Hsz, 4 * Hsz, 1, Csz, Csz, Hsz);

  moglstm_main<<<NBLK, NTHR, 0, stream>>>(x, (float*)d_out, ws);
}
